// TemporalSelfAttention_15006615734127
// MI455X (gfx1250) — matrix operand order repaired and checked by reading, not compiled, not run on hardware
//
#include <hip/hip_runtime.h>
#include <hip/hip_bf16.h>

typedef __attribute__((ext_vector_type(16))) _Float16 v16h;
typedef __attribute__((ext_vector_type(8)))  _Float16 v8h;
typedef __attribute__((ext_vector_type(4)))  _Float16 v4h;
typedef __attribute__((ext_vector_type(8)))  float    v8f;

// ---- problem constants (match reference) ----
constexpr int NQ    = 40000;   // 200*200
constexpr int EMB   = 256;
constexpr int HEADS = 8;
constexpr int DH    = 32;      // EMB/HEADS
constexpr int PTS   = 4;
constexpr int QUE   = 2;
constexpr int GW    = 200, GH = 200;
constexpr int NOFF  = 128;     // QUE*HEADS*1*PTS*2
constexpr int NAW   = 64;      // QUE*HEADS*1*PTS
constexpr int NCAT  = NOFF + NAW; // 192

// ---- GEMM tiling ----
constexpr int BM = 64, BN = 64, BK = 32, THREADS = 128;
constexpr int LDA = 40;            // padded LDS row stride (halfs): 80B = 20 banks
constexpr int CPT = (BM * BK) / (8 * THREADS); // 16B chunks per thread per tile = 2

// Flip to 0 if the async mnemonic is rejected by the assembler.
#define USE_ASYNC_LDS 1

__device__ __forceinline__ void wait_stage_done()
{
#if USE_ASYNC_LDS
  asm volatile("s_wait_asynccnt 0x0" ::: "memory");
#endif
}

// Copy a BM x BK f16 tile (row-contiguous, rowStride halfs) into padded LDS.
__device__ __forceinline__ void stage_tile(_Float16* __restrict__ dstLds,
                                           const _Float16* __restrict__ src0,
                                           int rowStrideHalfs, int tid)
{
  #pragma unroll
  for (int c = 0; c < CPT; ++c) {
    const int chunk = tid + c * THREADS;     // 256 chunks of 8 halfs
    const int row   = chunk >> 2;            // 4 chunks per 32-half row
    const int coff  = (chunk & 3) * 8;
    const _Float16* src = src0 + (size_t)row * rowStrideHalfs + coff;
    _Float16* dst = dstLds + row * LDA + coff;
#if USE_ASYNC_LDS
    asm volatile("global_load_async_to_lds_b128 %0, %1, off"
                 :: "v"((unsigned int)(unsigned long long)dst), "v"(src)
                 : "memory");
#else
    *(v8h*)dst = *(const v8h*)src;
#endif
  }
}

// MODE 0: vproj = value_h @ WvT^T + b_val        (M=80000,N=256,K=256) -> f16
// MODE 1: offaw = [value_h|q_h] @ WcT^T + bias    (M=40000,N=192,K=512) -> f32
// MODE 2: out   = attn_h @ WoT^T + b_out + query  (M=40000,N=256,K=256) -> f32
template<int MODE>
__global__ __launch_bounds__(THREADS)
void gemm_wmma(const _Float16* __restrict__ A0,  // value_h (M0,M1) / attn_h (M2)
               const _Float16* __restrict__ A1,  // q_h (M1 high-K half)
               const _Float16* __restrict__ WT,  // [N][K] f16 (pre-transposed)
               const float* __restrict__ bias0,
               const float* __restrict__ bias1,  // b_attn (M1)
               const float* __restrict__ resid,  // query identity (M2)
               _Float16* __restrict__ Ch,        // f16 out (M0)
               float* __restrict__ Cf,           // f32 out (M1/M2)
               int N, int K)
{
  __shared__ _Float16 As[2][BM * LDA];
  __shared__ _Float16 Bs[2][BN * LDA];   // [n][k]

  const int tid  = threadIdx.x;
  const int wave = tid >> 5;
  const int lane = tid & 31;
  const int lh   = lane & 15;
  const int hi   = lane >> 4;
  const int rowBase = blockIdx.x * BM;
  const int colBase = blockIdx.y * BN;

  auto stageA = [&](int buf, int kk) {
    const _Float16* src = (MODE == 1 && kk >= 256)
                        ? (A1 + (size_t)rowBase * 256 + (kk - 256))
                        : (A0 + (size_t)rowBase * 256 + kk);
    stage_tile(&As[buf][0], src, 256, tid);
  };
  auto stageB = [&](int buf, int kk) {
    stage_tile(&Bs[buf][0], WT + (size_t)colBase * K + kk, K, tid);
  };

  stageA(0, 0);
  stageB(0, 0);

  const int T = K / BK;
  v8f acc[4] = {};

  for (int t = 0; t < T; ++t) {
    wait_stage_done();        // this wave's async copies into buf t&1 complete
    __syncthreads();          // all waves' copies visible
    if (t + 1 < T) {          // overlap next tile DMA with this tile's WMMAs
      stageA((t + 1) & 1, (t + 1) * BK);
      stageB((t + 1) & 1, (t + 1) * BK);
    }

    const _Float16* as = &As[t & 1][0];
    const _Float16* bs = &Bs[t & 1][0];

    // A fragment (ISA 7.12.2 16-bit A 16x32): lanes 0-15: K{0..7,16..23}, 16-31: K{8..15,24..31}
    const _Float16* arow = as + (wave * 16 + lh) * LDA;
    const v8h a_lo = *(const v8h*)(arow + hi * 8);
    const v8h a_hi = *(const v8h*)(arow + hi * 8 + 16);
    v16h afrag;
    #pragma unroll
    for (int q = 0; q < 8; ++q) { afrag[q] = a_lo[q]; afrag[q + 8] = a_hi[q]; }

    #pragma unroll
    for (int j = 0; j < 4; ++j) {
      const _Float16* bcol = bs + (j * 16 + lh) * LDA + hi * 8; // K {0..7,16..23} | {8..15,24..31}
      const v8h b0 = *(const v8h*)(bcol);
      const v8h b1 = *(const v8h*)(bcol + 16);
      v16h bfrag;
      #pragma unroll
      for (int q = 0; q < 8; ++q) { bfrag[q] = b0[q]; bfrag[q + 8] = b1[q]; }
      acc[j] = __builtin_amdgcn_wmma_f32_16x16x32_f16(
          false, afrag, false, bfrag, (short)0, acc[j], false, false);
    }
  }

  // C/D layout: VGPR i holds M=i (lanes 0-15) / M=8+i (lanes 16-31); N=lane&15
  #pragma unroll
  for (int j = 0; j < 4; ++j) {
    const int gn = colBase + j * 16 + lh;
    #pragma unroll
    for (int i = 0; i < 8; ++i) {
      const int gm = rowBase + wave * 16 + hi * 8 + i;
      float val = acc[j][i];
      if (MODE == 0) {
        val += bias0[gn];
        Ch[(size_t)gm * N + gn] = (_Float16)val;
      } else if (MODE == 1) {
        val += (gn < NOFF) ? bias0[gn] : bias1[gn - NOFF];
        Cf[(size_t)gm * N + gn] = val;
      } else {
        val += bias0[gn] + resid[(size_t)gm * N + gn];
        Cf[(size_t)gm * N + gn] = val;
      }
    }
  }
}

// ---- prep: f32 -> f16 conversions (vectorized) ----
__global__ __launch_bounds__(256)
void cvt_f16(const float4* __restrict__ x, v4h* __restrict__ y, int n4)
{
  const int i = blockIdx.x * 256 + threadIdx.x;
  if (i >= n4) return;
  const float4 v = x[i];
  v4h h; h[0] = (_Float16)v.x; h[1] = (_Float16)v.y;
         h[2] = (_Float16)v.z; h[3] = (_Float16)v.w;
  y[i] = h;
}

__global__ __launch_bounds__(256)
void cvt_add_f16(const float4* __restrict__ a, const float4* __restrict__ b,
                 v4h* __restrict__ y, int n4)
{
  const int i = blockIdx.x * 256 + threadIdx.x;
  if (i >= n4) return;
  const float4 va = a[i], vb = b[i];
  v4h h; h[0] = (_Float16)(va.x + vb.x); h[1] = (_Float16)(va.y + vb.y);
         h[2] = (_Float16)(va.z + vb.z); h[3] = (_Float16)(va.w + vb.w);
  y[i] = h;
}

// Build transposed f16 weights: dst[n*K+k] = W[k][n]; optional concat split.
__global__ __launch_bounds__(256)
void build_wT(const float* __restrict__ W0, const float* __restrict__ W1,
              _Float16* __restrict__ dst, int K, int N, int split)
{
  const int idx = blockIdx.x * 256 + threadIdx.x;
  if (idx >= K * N) return;
  const int k = idx / N, n = idx % N;
  const float w = (split && n >= split) ? W1[(size_t)k * (N - split) + (n - split)]
                                        : W0[(size_t)k * (split ? split : N) + n];
  dst[(size_t)n * K + k] = (_Float16)w;
}

// ---- deformable sampling: one wave per (query, head); lane = channel ----
__device__ __forceinline__ float corner_val(const _Float16* vq, int xi, int yi, float wgt)
{
  if (xi < 0 || xi >= GW || yi < 0 || yi >= GH) return 0.f;
  return wgt * (float)vq[(size_t)(yi * GW + xi) * EMB];
}

__global__ __launch_bounds__(256)
void deform_sample(const _Float16* __restrict__ v,      // (QUE, NQ, EMB) f16
                   const float* __restrict__ offaw,     // (NQ, 192)
                   const float* __restrict__ refp,      // (QUE, NQ, 1, 2)
                   _Float16* __restrict__ out_attn)     // (NQ, EMB) f16
{
  const int wid  = (blockIdx.x * blockDim.x + threadIdx.x) >> 5;
  const int lane = threadIdx.x & 31;
  if (wid >= NQ * HEADS) return;
  const int q = wid >> 3;
  const int h = wid & 7;

  const float* oa = offaw + (size_t)q * NCAT;
  float acc = 0.f;

  #pragma unroll
  for (int qu = 0; qu < QUE; ++qu) {
    const float rx = refp[((size_t)qu * NQ + q) * 2 + 0];
    const float ry = refp[((size_t)qu * NQ + q) * 2 + 1];
    const int jb = (h * QUE + qu) * PTS;

    const float l0 = oa[NOFF + jb + 0], l1 = oa[NOFF + jb + 1];
    const float l2 = oa[NOFF + jb + 2], l3 = oa[NOFF + jb + 3];
    const float mx = fmaxf(fmaxf(l0, l1), fmaxf(l2, l3));
    const float e0 = __expf(l0 - mx), e1 = __expf(l1 - mx);
    const float e2 = __expf(l2 - mx), e3 = __expf(l3 - mx);
    const float inv = 1.f / (e0 + e1 + e2 + e3);
    const float wts[4] = { e0 * inv, e1 * inv, e2 * inv, e3 * inv };

    const _Float16* vq = v + (size_t)qu * NQ * EMB + h * DH + lane;

    #pragma unroll
    for (int p = 0; p < PTS; ++p) {
      const float ox = oa[(jb + p) * 2 + 0];
      const float oy = oa[(jb + p) * 2 + 1];
      const float x = rx * (float)GW + ox - 0.5f;   // grid_sample align_corners=False
      const float y = ry * (float)GH + oy - 0.5f;
      const float xf = floorf(x), yf = floorf(y);
      const int x0 = (int)xf, y0 = (int)yf;
      const float lx = x - xf, ly = y - yf;
      const float aw = wts[p];
      acc += aw * (corner_val(vq, x0,     y0,     (1.f - lx) * (1.f - ly)) +
                   corner_val(vq, x0 + 1, y0,     lx         * (1.f - ly)) +
                   corner_val(vq, x0,     y0 + 1, (1.f - lx) * ly        ) +
                   corner_val(vq, x0 + 1, y0 + 1, lx         * ly        ));
    }
  }
  out_attn[(size_t)q * EMB + h * DH + lane] = (_Float16)(0.5f * acc);
}

extern "C" void kernel_launch(void* const* d_in, const int* in_sizes, int n_in,
                              void* d_out, int out_size, void* d_ws, size_t ws_size,
                              hipStream_t stream)
{
  const float* query     = (const float*)d_in[0];
  const float* query_pos = (const float*)d_in[1];
  const float* value     = (const float*)d_in[2];
  const float* refp      = (const float*)d_in[3];
  // d_in[4] spatial_shapes: static [[200,200]]
  const float* W_off  = (const float*)d_in[5];
  const float* b_off  = (const float*)d_in[6];
  const float* W_attn = (const float*)d_in[7];
  const float* b_attn = (const float*)d_in[8];
  const float* W_val  = (const float*)d_in[9];
  const float* b_val  = (const float*)d_in[10];
  const float* W_out  = (const float*)d_in[11];
  const float* b_out  = (const float*)d_in[12];
  float* out = (float*)d_out;

  // ---- workspace layout (all 16B aligned) ----
  char* ws = (char*)d_ws;
  size_t o = 0;
  _Float16* value_h = (_Float16*)(ws + o); o += (size_t)QUE * NQ * EMB * 2;  // 40.96 MB
  _Float16* q_h     = (_Float16*)(ws + o); o += (size_t)NQ * EMB * 2;        // 20.48 MB
  _Float16* vproj_h = (_Float16*)(ws + o); o += (size_t)QUE * NQ * EMB * 2;  // 40.96 MB
  _Float16* attn_h  = (_Float16*)(ws + o); o += (size_t)NQ * EMB * 2;        // 20.48 MB
  float*    offaw   = (float*)   (ws + o); o += (size_t)NQ * NCAT * 4;       // 30.72 MB
  _Float16* WvT     = (_Float16*)(ws + o); o += (size_t)EMB * EMB * 2;
  _Float16* WcT     = (_Float16*)(ws + o); o += (size_t)2 * EMB * NCAT * 2;
  _Float16* WoT     = (_Float16*)(ws + o); o += (size_t)EMB * EMB * 2;       // ~154 MB total

  // ---- prep: lower operands to f16 (weights transposed to [N][K]) ----
  {
    const int n4v = (QUE * NQ * EMB) / 4;
    cvt_f16<<<(n4v + 255) / 256, 256, 0, stream>>>((const float4*)value, (v4h*)value_h, n4v);
    const int n4q = (NQ * EMB) / 4;
    cvt_add_f16<<<(n4q + 255) / 256, 256, 0, stream>>>((const float4*)query,
                                                       (const float4*)query_pos,
                                                       (v4h*)q_h, n4q);
    build_wT<<<(EMB * EMB + 255) / 256, 256, 0, stream>>>(W_val, nullptr, WvT, EMB, EMB, 0);
    build_wT<<<(2 * EMB * NCAT + 255) / 256, 256, 0, stream>>>(W_off, W_attn, WcT,
                                                               2 * EMB, NCAT, NOFF);
    build_wT<<<(EMB * EMB + 255) / 256, 256, 0, stream>>>(W_out, nullptr, WoT, EMB, EMB, 0);
  }

  // 1) value projection (f16 out, consumed by gather)
  gemm_wmma<0><<<dim3((QUE * NQ) / BM, EMB / BN), THREADS, 0, stream>>>(
      value_h, nullptr, WvT, b_val, nullptr, nullptr, vproj_h, nullptr, EMB, EMB);

  // 2) fused offsets + attention logits: q2 @ [W_off|W_attn] + bias
  gemm_wmma<1><<<dim3(NQ / BM, NCAT / BN), THREADS, 0, stream>>>(
      value_h, q_h, WcT, b_off, b_attn, nullptr, nullptr, offaw, NCAT, 2 * EMB);

  // 3) softmax + bilinear deformable sampling + queue mean
  deform_sample<<<(NQ * HEADS) / 8, 256, 0, stream>>>(vproj_h, offaw, refp, attn_h);

  // 4) output projection + bias + residual
  gemm_wmma<2><<<dim3(NQ / BM, EMB / BN), THREADS, 0, stream>>>(
      attn_h, nullptr, WoT, b_out, nullptr, query, nullptr, out, EMB, EMB);
}
